// BlockAttentionPairBias_59115929862885
// MI455X (gfx1250) — matrix operand order repaired and checked by reading, not compiled, not run on hardware
//
#include <hip/hip_runtime.h>
#include <hip/hip_bf16.h>

// ---------------------------------------------------------------------------
// BlockAttentionPairBias for MI455X (gfx1250, wave32, WMMA 16x16x32 f16)
// B=2, N=1024, C_S=768, C_Z=64, H=16, DH=48 (padded to 64 in ws buffers)
// ---------------------------------------------------------------------------

#define BATCH 2
#define NSEQ  1024
#define CS    768
#define CZ    64
#define NH    16
#define DH    48
#define DHP   64          // padded head dim
#define INF_V 100000.0f
#define KP    384         // K pairs per 768-deep GEMM
#define NT_PROJ 192       // 3072/16 col tiles of fused projection
#define NT_OUT  48        // 768/16 col tiles of output projection

typedef __attribute__((ext_vector_type(16))) _Float16 v16h;
typedef __attribute__((ext_vector_type(8)))  float    v8f;

union Frag { v16h v; unsigned int u[8]; };

__device__ __forceinline__ unsigned int pack2f16(float a, float b) {
    union { _Float16 h[2]; unsigned int u; } p;
    p.h[0] = (_Float16)a; p.h[1] = (_Float16)b; return p.u;
}

__device__ __forceinline__ v8f wmma_f16(const Frag& a, const Frag& b, v8f c) {
    // D = A(16x32 f16) * B(32x16 f16) + C(16x16 f32)
    return __builtin_amdgcn_wmma_f32_16x16x32_f16(
        /*neg_a=*/false, a.v, /*neg_b=*/false, b.v,
        /*c_mod=*/(short)0, c, /*reuse_a=*/false, /*reuse_b=*/false);
}

// A-fragment: VGPR r of lane (half = lane>>4) holds K pair starting at:
__device__ __forceinline__ int kA_of(int r, int half) {
    return (half << 4) | (r << 1);
}
// B-fragment: VGPR r of lane (half = lane>>4) holds K pair starting at:
__device__ __forceinline__ int kB_of(int r, int half) {
    return (half << 4) | (r << 1);
}

// ---------------------------------------------------------------------------
// 0) zero fill (for the d = 48..63 padding of Q/K/V)
// ---------------------------------------------------------------------------
__global__ void k_zero(uint4* p, long n4) {
    for (long i = (long)blockIdx.x * blockDim.x + threadIdx.x; i < n4;
         i += (long)gridDim.x * blockDim.x)
        p[i] = make_uint4(0u, 0u, 0u, 0u);
}

// ---------------------------------------------------------------------------
// 0b) pre-pack all GEMM weights (fp32, row-major) into f16 B-fragment layout:
//     uint wp[tile][kpair][lane16] = pack(W[2k][n], W[2k+1][n]).
//     Tiles 0..191: [w_q | w_kv | w_gate]; tiles 192..239: w_out.
//     One-time ~6 MB; makes the hot GEMM loops pure load+WMMA.
// ---------------------------------------------------------------------------
__global__ __launch_bounds__(256) void k_packw(const float* __restrict__ w_q,
                                               const float* __restrict__ w_kv,
                                               const float* __restrict__ w_gate,
                                               const float* __restrict__ w_out,
                                               unsigned int* __restrict__ wp) {
    const long idx = (long)blockIdx.x * 256 + threadIdx.x; // 240*384*16 total
    const int ln = (int)(idx & 15);
    const int kp = (int)((idx >> 4) % KP);
    const int nt = (int)(idx / (KP * 16));
    const int k  = kp << 1;
    const float* W; int ldn, col;
    const int nbase = nt << 4;
    if (nt < NT_PROJ) {
        if (nbase < 768)       { W = w_q;    ldn = 768;  col = nbase + ln; }
        else if (nbase < 2304) { W = w_kv;   ldn = 1536; col = nbase - 768 + ln; }
        else                   { W = w_gate; ldn = 768;  col = nbase - 2304 + ln; }
    } else {
        W = w_out; ldn = 768; col = ((nt - NT_PROJ) << 4) + ln;
    }
    wp[idx] = pack2f16(W[(size_t)k * ldn + col], W[(size_t)(k + 1) * ldn + col]);
}

// ---------------------------------------------------------------------------
// 1) LayerNorm(node_embed) -> s (f16). One 256-thread block per row of 768.
// ---------------------------------------------------------------------------
__global__ __launch_bounds__(256) void k_ln_node(const float* __restrict__ x,
                                                 const float* __restrict__ gam,
                                                 const float* __restrict__ bet,
                                                 _Float16* __restrict__ s) {
    __shared__ float r1[256], r2[256];
    const int row = blockIdx.x;                 // 0..B*N-1
    const float* xr = x + (size_t)row * CS;
    const int t = threadIdx.x;
    float a0 = xr[t], a1 = xr[t + 256], a2 = xr[t + 512];
    r1[t] = a0 + a1 + a2;
    r2[t] = a0 * a0 + a1 * a1 + a2 * a2;
    __syncthreads();
    for (int o = 128; o > 0; o >>= 1) {
        if (t < o) { r1[t] += r1[t + o]; r2[t] += r2[t + o]; }
        __syncthreads();
    }
    const float mu  = r1[0] * (1.0f / CS);
    const float var = r2[0] * (1.0f / CS) - mu * mu;
    const float rs  = rsqrtf(var + 1e-5f);
    _Float16* sr = s + (size_t)row * CS;
    sr[t]       = (_Float16)((a0 - mu) * rs * gam[t]       + bet[t]);
    sr[t + 256] = (_Float16)((a1 - mu) * rs * gam[t + 256] + bet[t + 256]);
    sr[t + 512] = (_Float16)((a2 - mu) * rs * gam[t + 512] + bet[t + 512]);
}

// ---------------------------------------------------------------------------
// 2) Fused projections: s(2048x768 f16) @ packed[w_q|w_kv|w_gate] (768x3072)
//    -> Q/K: (B,H,N,DHP) f16   V: (B,H,DHP,N) f16 (transposed)  gate: fp32
//    8 waves/block, one 16x16 output tile per wave, K-loop of 24 WMMAs.
// ---------------------------------------------------------------------------
__global__ __launch_bounds__(256) void k_proj(const _Float16* __restrict__ s,
                                              const unsigned int* __restrict__ wpack,
                                              _Float16* __restrict__ qb,
                                              _Float16* __restrict__ kb,
                                              _Float16* __restrict__ vb,
                                              float* __restrict__ gateb) {
    const int lane = threadIdx.x & 31, wave = threadIdx.x >> 5;
    const int tile = blockIdx.x * 8 + wave;     // 128 * 192 tiles
    const int mt = tile / NT_PROJ, nt = tile % NT_PROJ;
    const int m0 = mt << 4, nbase = nt << 4;
    const int half = lane >> 4, ln = lane & 15;

    const unsigned int* wp = wpack + (size_t)nt * (KP * 16);
    const unsigned int* arow = (const unsigned int*)(s + (size_t)(m0 + ln) * CS);

    v8f c = {};
#pragma unroll 2
    for (int kk = 0; kk < CS; kk += 32) {
        Frag A, Bf;
        const int kpb = kk >> 1;
#pragma unroll
        for (int r = 0; r < 8; ++r)
            A.u[r] = arow[(kk + kA_of(r, half)) >> 1];
#pragma unroll
        for (int r = 0; r < 8; ++r)
            Bf.u[r] = wp[((kpb + (half << 3) + r) << 4) + ln];
        c = wmma_f16(A, Bf, c);
    }
#pragma unroll
    for (int r = 0; r < 8; ++r) {
        const int M = r + (half << 3);
        const int m = m0 + M, b = m >> 10, i = m & (NSEQ - 1);
        const int n = nbase + ln;
        const float val = c[r];
        if (nbase < 768) {
            const int h = n / DH, d = n % DH;
            qb[(((size_t)(b * NH + h)) * NSEQ + i) * DHP + d] = (_Float16)val;
        } else if (nbase < 1536) {
            const int c2 = n - 768, h = c2 / DH, d = c2 % DH;
            kb[(((size_t)(b * NH + h)) * NSEQ + i) * DHP + d] = (_Float16)val;
        } else if (nbase < 2304) {
            const int c3 = n - 1536, h = c3 / DH, d = c3 % DH;
            vb[(((size_t)(b * NH + h)) * DHP + d) * NSEQ + i] = (_Float16)val;
        } else {
            const int c4 = n - 2304;
            gateb[(size_t)m * CS + c4] = 1.0f / (1.0f + __expf(-val));
        }
    }
}

// ---------------------------------------------------------------------------
// 3) Pair bias: LN(edge row of 64) @ w_bias(64x16) -> bias (B,N,N,H) f16.
//    One wave handles 16 consecutive (i,j) rows: coalesced float4 loads,
//    pair-lane LN, restage through LDS into A-fragments, 2 WMMAs (K=64).
// ---------------------------------------------------------------------------
__global__ __launch_bounds__(256) void k_bias(const float* __restrict__ edge,
                                              const float* __restrict__ gam,
                                              const float* __restrict__ bet,
                                              const float* __restrict__ w_bias,
                                              _Float16* __restrict__ biasb) {
    __shared__ float sg[CZ], sb[CZ];
    __shared__ _Float16 tile[8][16][CZ];        // 16 KB
    if (threadIdx.x < CZ) { sg[threadIdx.x] = gam[threadIdx.x];
                            sb[threadIdx.x] = bet[threadIdx.x]; }
    __syncthreads();

    const int lane = threadIdx.x & 31, wave = threadIdx.x >> 5;
    const size_t r0 = ((size_t)blockIdx.x * 8 + wave) * 16; // first edge row
    const int b   = (int)(r0 >> 20);
    const int rem = (int)(r0 & ((1u << 20) - 1));
    const int i = rem >> 10, j0 = rem & (NSEQ - 1);

    // lane pair (2k,2k+1) covers row k: each lane loads 32 contiguous floats
    const int row = lane >> 1, chalf = lane & 1;
    const float* ep = edge + (r0 + row) * CZ + chalf * 32;
    float e[32];
#pragma unroll
    for (int tt = 0; tt < 32; tt += 4) {
        const float4 f = *(const float4*)(ep + tt);
        e[tt] = f.x; e[tt + 1] = f.y; e[tt + 2] = f.z; e[tt + 3] = f.w;
    }
    float sum = 0.f, sq = 0.f;
#pragma unroll
    for (int tt = 0; tt < 32; ++tt) { sum += e[tt]; sq += e[tt] * e[tt]; }
    sum += __shfl_xor(sum, 1, 32);
    sq  += __shfl_xor(sq, 1, 32);
    const float mu  = sum * (1.0f / CZ);
    const float var = sq * (1.0f / CZ) - mu * mu;
    const float rs  = rsqrtf(var + 1e-5f);
#pragma unroll
    for (int tt = 0; tt < 32; ++tt) {
        const int cidx = chalf * 32 + tt;
        tile[wave][row][cidx] = (_Float16)((e[tt] - mu) * rs * sg[cidx] + sb[cidx]);
    }
    // gather A fragments from LDS (same-wave DS ordering handled by compiler)
    const int half = lane >> 4, ln = lane & 15;
    const unsigned int* tp = (const unsigned int*)&tile[wave][ln][0];
    Frag A0, A1, B0, B1;
#pragma unroll
    for (int r = 0; r < 8; ++r) {
        const int ka = kA_of(r, half);
        A0.u[r] = tp[ka >> 1];
        A1.u[r] = tp[(ka + 32) >> 1];
    }
#pragma unroll
    for (int r = 0; r < 8; ++r) {
        const int k = kB_of(r, half);
        B0.u[r] = pack2f16(w_bias[(size_t)k * NH + ln],
                           w_bias[(size_t)(k + 1) * NH + ln]);
        B1.u[r] = pack2f16(w_bias[(size_t)(k + 32) * NH + ln],
                           w_bias[(size_t)(k + 33) * NH + ln]);
    }
    v8f c = {};
    c = wmma_f16(A0, B0, c);
    c = wmma_f16(A1, B1, c);
#pragma unroll
    for (int r = 0; r < 8; ++r) {
        const int M = r + (half << 3);          // row -> j offset
        biasb[(((size_t)b * NSEQ + i) * NSEQ + (j0 + M)) * NH + ln] = (_Float16)c[r];
    }
}

// ---------------------------------------------------------------------------
// 4) Flash attention: block = (b, 16-row i-tile), 16 waves = one head each.
//    Per j-step of 32: 4 logit WMMAs (K=d), bias+mask, streaming softmax,
//    p transposed through LDS, 3 P@V WMMAs (d tiles). Gated f16 epilogue.
//    global_prefetch for next step's K/V/bias rows (streaming, BW-bound).
// ---------------------------------------------------------------------------
__global__ __launch_bounds__(512) void k_attn(const _Float16* __restrict__ qb,
                                              const _Float16* __restrict__ kb,
                                              const _Float16* __restrict__ vb,
                                              const _Float16* __restrict__ biasb,
                                              const int* __restrict__ mask,
                                              const float* __restrict__ gateb,
                                              _Float16* __restrict__ outp) {
    __shared__ _Float16 pbuf[NH][16][32];       // 16 KB, one 16x32 tile / wave
    const int lane = threadIdx.x & 31, h = threadIdx.x >> 5;
    const int b = blockIdx.x >> 6, i0 = (blockIdx.x & 63) << 4;
    const int half = lane >> 4, ln = lane & 15;
    const size_t bh = (size_t)(b * NH + h);

    // Q fragments (K = d, padded 48->64 with zeros)
    Frag QA0, QA1;
    {
        const unsigned int* qp =
            (const unsigned int*)(qb + (bh * NSEQ + i0 + ln) * DHP);
#pragma unroll
        for (int r = 0; r < 8; ++r) {
            const int ka = kA_of(r, half);
            QA0.u[r] = qp[ka >> 1];
            QA1.u[r] = qp[(ka + 32) >> 1];
        }
    }
    float mrun[8], lrun[8], mi[8];
    v8f acc[3] = {};
#pragma unroll
    for (int r = 0; r < 8; ++r) {
        mrun[r] = -1e30f; lrun[r] = 0.f;
        mi[r] = (float)mask[b * NSEQ + i0 + r + (half << 3)];
    }
    const float invs = 0.14433756729740643f;    // 1/sqrt(48)

    for (int j0 = 0; j0 < NSEQ; j0 += 32) {
        // ---- prefetch next j-step's streams (emits global_prefetch_b8)
        if (j0 + 32 < NSEQ) {
            const int jn = j0 + 32;
            __builtin_prefetch(kb + (bh * NSEQ + jn + lane) * DHP, 0, 3);
            __builtin_prefetch(vb + (bh * DHP + lane) * NSEQ + jn, 0, 3);
            __builtin_prefetch(
                biasb + (((size_t)b * NSEQ + i0 + ln) * NSEQ + jn) * NH, 0, 3);
        }
        // ---- logits for two 16-wide j half-tiles
        v8f c0 = {}, c1 = {};
#pragma unroll
        for (int jt = 0; jt < 2; ++jt) {
            const int j = j0 + jt * 16 + ln;
            const unsigned int* kp =
                (const unsigned int*)(kb + (bh * NSEQ + j) * DHP);
            Frag B0, B1;
#pragma unroll
            for (int r = 0; r < 8; ++r) {
                const int k = kB_of(r, half);
                B0.u[r] = kp[k >> 1];
                B1.u[r] = kp[(k + 32) >> 1];
            }
            v8f cc = jt ? c1 : c0;
            cc = wmma_f16(QA0, B0, cc);
            cc = wmma_f16(QA1, B1, cc);
            if (jt) c1 = cc; else c0 = cc;
        }
        // ---- scale + pair bias + mask
#pragma unroll
        for (int jt = 0; jt < 2; ++jt) {
            const int j = j0 + jt * 16 + ln;
            const float mj = (float)mask[b * NSEQ + j];
#pragma unroll
            for (int r = 0; r < 8; ++r) {
                const int i = i0 + r + (half << 3);
                const float bv = (float)biasb[(((size_t)b * NSEQ + i) * NSEQ + j) * NH + h];
                const float pen = (mi[r] * mj - 1.0f) * INF_V;
                if (jt) c1[r] = c1[r] * invs + bv + pen;
                else    c0[r] = c0[r] * invs + bv + pen;
            }
        }
        // ---- streaming softmax (row = (r, half); reduce across 16 lanes)
#pragma unroll
        for (int r = 0; r < 8; ++r) {
            float t = fmaxf(c0[r], c1[r]);
            t = fmaxf(t, __shfl_xor(t, 1, 16));
            t = fmaxf(t, __shfl_xor(t, 2, 16));
            t = fmaxf(t, __shfl_xor(t, 4, 16));
            t = fmaxf(t, __shfl_xor(t, 8, 16));
            const float mnew = fmaxf(mrun[r], t);
            const float es = __expf(mrun[r] - mnew);
            mrun[r] = mnew;
            const float p0 = __expf(c0[r] - mnew);
            const float p1 = __expf(c1[r] - mnew);
            c0[r] = p0; c1[r] = p1;
            float rsum = p0 + p1;
            rsum += __shfl_xor(rsum, 1, 16);
            rsum += __shfl_xor(rsum, 2, 16);
            rsum += __shfl_xor(rsum, 4, 16);
            rsum += __shfl_xor(rsum, 8, 16);
            lrun[r] = lrun[r] * es + rsum;
            acc[0][r] *= es; acc[1][r] *= es; acc[2][r] *= es;
        }
        // ---- transpose p (C layout -> A layout) via this wave's LDS tile
#pragma unroll
        for (int r = 0; r < 8; ++r) {
            const int M = r + (half << 3);
            pbuf[h][M][ln]      = (_Float16)c0[r];
            pbuf[h][M][16 + ln] = (_Float16)c1[r];
        }
        Frag PA;
        {
            const unsigned int* pp = (const unsigned int*)&pbuf[h][ln][0];
#pragma unroll
            for (int r = 0; r < 8; ++r) PA.u[r] = pp[kA_of(r, half) >> 1];
        }
        // ---- P @ V over three 16-wide d tiles (V stored transposed)
#pragma unroll
        for (int dt = 0; dt < 3; ++dt) {
            const int d = dt * 16 + ln;
            const unsigned int* vp =
                (const unsigned int*)(vb + (bh * DHP + d) * NSEQ + j0 + (half << 4));
            Frag VB;
#pragma unroll
            for (int r = 0; r < 8; ++r) VB.u[r] = vp[r];
            acc[dt] = wmma_f16(PA, VB, acc[dt]);
        }
    }
    // ---- epilogue: normalize, gate, write f16 pre-projection activation
#pragma unroll
    for (int dt = 0; dt < 3; ++dt) {
#pragma unroll
        for (int r = 0; r < 8; ++r) {
            const int i = i0 + r + (half << 3);
            const int ccol = h * DH + dt * 16 + ln;
            const size_t idx = ((size_t)b * NSEQ + i) * CS + ccol;
            float o = acc[dt][r] / lrun[r];
            o *= gateb[idx];
            outp[idx] = (_Float16)o;
        }
    }
}

// ---------------------------------------------------------------------------
// 5) Output projection: (out*gate)(2048x768 f16) @ packed w_out -> fp32
// ---------------------------------------------------------------------------
__global__ __launch_bounds__(256) void k_out(const _Float16* __restrict__ a,
                                             const unsigned int* __restrict__ wpack,
                                             float* __restrict__ out) {
    const int lane = threadIdx.x & 31, wave = threadIdx.x >> 5;
    const int tile = blockIdx.x * 8 + wave;     // 128 * 48 tiles
    const int mt = tile / NT_OUT, nt = tile % NT_OUT;
    const int m0 = mt << 4, n0 = nt << 4;
    const int half = lane >> 4, ln = lane & 15;

    const unsigned int* wp = wpack + (size_t)(NT_PROJ + nt) * (KP * 16);
    const unsigned int* arow = (const unsigned int*)(a + (size_t)(m0 + ln) * CS);

    v8f c = {};
#pragma unroll 2
    for (int kk = 0; kk < CS; kk += 32) {
        Frag A, Bf;
        const int kpb = kk >> 1;
#pragma unroll
        for (int r = 0; r < 8; ++r) A.u[r] = arow[(kk + kA_of(r, half)) >> 1];
#pragma unroll
        for (int r = 0; r < 8; ++r)
            Bf.u[r] = wp[((kpb + (half << 3) + r) << 4) + ln];
        c = wmma_f16(A, Bf, c);
    }
#pragma unroll
    for (int r = 0; r < 8; ++r) {
        const int M = r + (half << 3);
        out[(size_t)(m0 + M) * CS + n0 + ln] = c[r];
    }
}

// ---------------------------------------------------------------------------
// launch
// ---------------------------------------------------------------------------
extern "C" void kernel_launch(void* const* d_in, const int* in_sizes, int n_in,
                              void* d_out, int out_size, void* d_ws, size_t ws_size,
                              hipStream_t stream) {
    const float* node   = (const float*)d_in[0];
    const float* edge   = (const float*)d_in[1];
    const int*   nmask  = (const int*)  d_in[2];
    const float* ln_g   = (const float*)d_in[3];
    const float* ln_b   = (const float*)d_in[4];
    const float* w_q    = (const float*)d_in[5];
    const float* w_kv   = (const float*)d_in[6];
    const float* bln_g  = (const float*)d_in[7];
    const float* bln_b  = (const float*)d_in[8];
    const float* w_bias = (const float*)d_in[9];
    const float* w_gate = (const float*)d_in[10];
    const float* w_out  = (const float*)d_in[11];
    float* out = (float*)d_out;

    // workspace carve-up
    char* p = (char*)d_ws;
    const size_t sz_s     = (size_t)BATCH * NSEQ * CS * sizeof(_Float16);        // 3 MB
    const size_t sz_head  = (size_t)BATCH * NH * NSEQ * DHP * sizeof(_Float16);  // 4 MB
    const size_t sz_gate  = (size_t)BATCH * NSEQ * CS * sizeof(float);           // 6 MB
    const size_t sz_bias  = (size_t)BATCH * NSEQ * NSEQ * NH * sizeof(_Float16); // 64 MB
    const size_t sz_wpack = (size_t)(NT_PROJ + NT_OUT) * KP * 16 * sizeof(unsigned int); // ~5.6 MB
    _Float16* s_f16  = (_Float16*)p;           p += sz_s;
    _Float16* qb     = (_Float16*)p;           p += sz_head;
    _Float16* kb     = (_Float16*)p;           p += sz_head;
    _Float16* vb     = (_Float16*)p;           p += sz_head;
    float*    gateb  = (float*)p;              p += sz_gate;
    _Float16* biasb  = (_Float16*)p;           p += sz_bias;
    _Float16* outp   = (_Float16*)p;           p += sz_s;
    unsigned int* wpack = (unsigned int*)p;    p += sz_wpack;

    // 0) zero d-padding of Q/K/V (3 x 4 MB)
    const long n4 = (long)(3 * sz_head / sizeof(uint4));
    k_zero<<<1024, 256, 0, stream>>>((uint4*)qb, n4);

    // 0b) one-time weight pre-pack into f16 B-fragment layout
    const long npack = (long)(NT_PROJ + NT_OUT) * KP * 16;
    k_packw<<<(int)(npack / 256), 256, 0, stream>>>(w_q, w_kv, w_gate, w_out, wpack);

    // 1) node LayerNorm
    k_ln_node<<<BATCH * NSEQ, 256, 0, stream>>>(node, ln_g, ln_b, s_f16);

    // 2) fused Q/KV/gate projections (M=2048, N=3072, K=768)
    k_proj<<<(128 * NT_PROJ) / 8, 256, 0, stream>>>(s_f16, wpack, qb, kb, vb, gateb);

    // 3) pair bias from edge_embed (single streaming pass over 512 MB)
    k_bias<<<(BATCH * NSEQ * NSEQ / 16) / 8, 256, 0, stream>>>(
        edge, bln_g, bln_b, w_bias, biasb);

    // 4) flash attention, one block per (b, i-tile), 16 waves = 16 heads
    k_attn<<<BATCH * (NSEQ / 16), 512, 0, stream>>>(
        qb, kb, vb, biasb, nmask, gateb, outp);

    // 5) output projection -> fp32 result
    k_out<<<(128 * NT_OUT) / 8, 256, 0, stream>>>(outp, wpack, out);
}
